// Log_Rbm_1013612282133
// MI455X (gfx1250) — compile-verified, matrix operand order checked by reading, not run on hardware
//
#include <hip/hip_runtime.h>
#include <hip/hip_bf16.h>

typedef __bf16 bf16;
typedef __attribute__((ext_vector_type(16))) __bf16 v16bf;
typedef __attribute__((ext_vector_type(8)))  __bf16 v8bf;
typedef __attribute__((ext_vector_type(8)))  float  v8f;

static constexpr int BB   = 256;   // batch
static constexpr int DIN  = 1024;
static constexpr int RR   = 512;
static constexpr int DOUT = 1024;

// ---------------------------------------------------------------------------
// Kernel 0: fp32 -> bf16 bulk convert (8 elements / thread, b128-in / b128-out
// wide enough: 32B load, 16B store).
// ---------------------------------------------------------------------------
__global__ __launch_bounds__(256) void k_f32_to_bf16(const float* __restrict__ src,
                                                     bf16* __restrict__ dst,
                                                     int n8) {
  const int i = blockIdx.x * blockDim.x + threadIdx.x;
  if (i < n8) {
    v8f f = ((const v8f*)src)[i];
    v8bf o;
#pragma unroll
    for (int j = 0; j < 8; ++j) o[j] = (bf16)f[j];
    ((v8bf*)dst)[i] = o;
  }
}

// ---------------------------------------------------------------------------
// Kernel 1: Ev = exp(xb @ Wb^T), bf16 in / bf16 out (BB x RR).
// One 16x16 output tile per wave; pure b128 bf16 loads -> WMMA, no converts.
// B-matrix column n == W row n (contiguous).
// ---------------------------------------------------------------------------
__global__ __launch_bounds__(256) void k_gemm1_exp(const bf16* __restrict__ xb,
                                                   const bf16* __restrict__ Wb,
                                                   bf16* __restrict__ Ev) {
  const int lane = threadIdx.x & 31;
  const int wave = threadIdx.x >> 5;
  const int tile = blockIdx.x * 8 + wave;      // 512 tiles = 16 M-tiles * 32 N-tiles
  const int tm   = (tile >> 5) << 4;
  const int tn   = (tile & 31) << 4;
  const int half = lane >> 4;                  // 0: K runs {0..7,16..23}, 1: {8..15,24..31}
  const int l16  = lane & 15;

  const bf16* arow = xb + (tm + l16) * DIN;    // A row m
  const bf16* brow = Wb + (tn + l16) * DIN;    // B column n = W row n

  v8f acc = {};
#pragma unroll 4
  for (int k0 = 0; k0 < DIN; k0 += 32) {
    const int ka = k0 + half * 8;
    const int kb = k0 + 16 + half * 8;
    v8bf alo = *(const v8bf*)(arow + ka);
    v8bf ahi = *(const v8bf*)(arow + kb);
    v8bf blo = *(const v8bf*)(brow + ka);
    v8bf bhi = *(const v8bf*)(brow + kb);
    v16bf a = __builtin_shufflevector(alo, ahi, 0,1,2,3,4,5,6,7,8,9,10,11,12,13,14,15);
    v16bf b = __builtin_shufflevector(blo, bhi, 0,1,2,3,4,5,6,7,8,9,10,11,12,13,14,15);
    acc = __builtin_amdgcn_wmma_f32_16x16x32_bf16(false, a, false, b,
                                                  (short)0, acc, false, false);
  }

  // C/D layout: acc[e] -> row tm + half*8 + e, col tn + l16
  bf16* out = Ev + (tm + half * 8) * RR + tn + l16;
#pragma unroll
  for (int e = 0; e < 8; ++e) out[e * RR] = (bf16)__expf(acc[e]);
}

// ---------------------------------------------------------------------------
// Kernel 2: EuT[d][r] = exp(u[r][d]), bf16 (DOUT x RR). LDS tile transpose so
// both the global read and the global write are coalesced.
// ---------------------------------------------------------------------------
__global__ __launch_bounds__(256) void k_exp_transpose(const float* __restrict__ u,
                                                       bf16* __restrict__ EuT) {
  __shared__ float tile[32][33];
  const int tx = threadIdx.x;        // 0..31
  const int ty = threadIdx.y;        // 0..7
  const int d0 = blockIdx.x * 32;
  const int r0 = blockIdx.y * 32;

#pragma unroll
  for (int j = 0; j < 32; j += 8)
    tile[ty + j][tx] = u[(r0 + ty + j) * DOUT + d0 + tx];

  __syncthreads();

#pragma unroll
  for (int j = 0; j < 32; j += 8)
    EuT[(d0 + ty + j) * RR + r0 + tx] = (bf16)__expf(tile[tx][ty + j]);
}

// ---------------------------------------------------------------------------
// Kernel 3: y = Ev @ EuT^T + RR.  A rows and B columns (EuT rows) are both
// contiguous bf16 -> pure b128 loads straight into WMMA fragments.
// ---------------------------------------------------------------------------
__global__ __launch_bounds__(256) void k_gemm2(const bf16* __restrict__ Ev,
                                               const bf16* __restrict__ EuT,
                                               float* __restrict__ y) {
  const int lane = threadIdx.x & 31;
  const int wave = threadIdx.x >> 5;
  const int tile = blockIdx.x * 8 + wave;      // 1024 tiles = 16 M-tiles * 64 N-tiles
  const int tm   = (tile >> 6) << 4;
  const int tn   = (tile & 63) << 4;
  const int half = lane >> 4;
  const int l16  = lane & 15;

  const bf16* arow = Ev  + (tm + l16) * RR;
  const bf16* brow = EuT + (tn + l16) * RR;

  v8f acc = {};
#pragma unroll 4
  for (int k0 = 0; k0 < RR; k0 += 32) {
    const int ka = k0 + half * 8;
    const int kb = k0 + 16 + half * 8;
    v8bf alo = *(const v8bf*)(arow + ka);
    v8bf ahi = *(const v8bf*)(arow + kb);
    v8bf blo = *(const v8bf*)(brow + ka);
    v8bf bhi = *(const v8bf*)(brow + kb);
    v16bf a = __builtin_shufflevector(alo, ahi, 0,1,2,3,4,5,6,7,8,9,10,11,12,13,14,15);
    v16bf b = __builtin_shufflevector(blo, bhi, 0,1,2,3,4,5,6,7,8,9,10,11,12,13,14,15);
    acc = __builtin_amdgcn_wmma_f32_16x16x32_bf16(false, a, false, b,
                                                  (short)0, acc, false, false);
  }

  float* out = y + (tm + half * 8) * DOUT + tn + l16;
#pragma unroll
  for (int e = 0; e < 8; ++e) out[e * DOUT] = acc[e] + (float)RR;
}

// ---------------------------------------------------------------------------
extern "C" void kernel_launch(void* const* d_in, const int* in_sizes, int n_in,
                              void* d_out, int out_size, void* d_ws, size_t ws_size,
                              hipStream_t stream) {
  const float* x = (const float*)d_in[0];   // (256, 1024)
  const float* W = (const float*)d_in[1];   // (512, 1024)
  const float* u = (const float*)d_in[2];   // (512, 1024)
  float* y = (float*)d_out;                 // (256, 1024)

  char* ws = (char*)d_ws;
  bf16* Ev  = (bf16*)(ws);                                   // 256*512   = 256 KB
  bf16* EuT = (bf16*)(ws + (size_t)BB * RR * 2);             // 1024*512  = 1 MB
  bf16* xb  = (bf16*)(ws + (size_t)(BB * RR + DOUT * RR) * 2);        // 512 KB
  bf16* Wb  = (bf16*)(ws + (size_t)(BB * RR + DOUT * RR + BB * DIN) * 2); // 1 MB

  const int xn8 = BB * DIN / 8;   // 32768
  const int wn8 = RR * DIN / 8;   // 65536
  k_f32_to_bf16<<<xn8 / 256, 256, 0, stream>>>(x, xb, xn8);
  k_f32_to_bf16<<<wn8 / 256, 256, 0, stream>>>(W, Wb, wn8);

  k_gemm1_exp<<<(BB / 16) * (RR / 16) / 8, 256, 0, stream>>>(xb, Wb, Ev);
  k_exp_transpose<<<dim3(DOUT / 32, RR / 32), dim3(32, 8), 0, stream>>>(u, EuT);
  k_gemm2<<<(BB / 16) * (DOUT / 16) / 8, 256, 0, stream>>>(Ev, EuT, y);
}
